// Decoder_37623913513321
// MI455X (gfx1250) — compile-verified, matrix operand order checked by reading, not run on hardware
//
#include <hip/hip_runtime.h>
#include <math.h>

// Problem constants
#define Bsz 128
#define VN 49
#define VD 2048
#define Ee 512
#define Hh 1024
#define WV 10000
#define Ll 20
#define KCAT (VD + Ee)   // 2560
#define KX (Ee + Hh)     // 1536 (emb || h)
#define G4H (4 * Hh)     // 4096

typedef __bf16 bf16_t;
typedef __attribute__((ext_vector_type(16))) __bf16 v16bf;
typedef __attribute__((ext_vector_type(8)))  float  v8f;

union FragBF {
    v16bf v;
    uint4 q[2];
};

__device__ __forceinline__ float sigm(float x) { return 1.0f / (1.0f + __expf(-x)); }

__device__ __forceinline__ void frag_load(FragBF& f, const bf16_t* __restrict__ p) {
    f.q[0] = *(const uint4*)p;
    f.q[1] = *(const uint4*)(p + 16);
}
__device__ __forceinline__ void frag_zero(FragBF& f) {
    f.q[0] = make_uint4(0u, 0u, 0u, 0u);
    f.q[1] = make_uint4(0u, 0u, 0u, 0u);
}

// ---------------------------------------------------------------------------
// Generic bf16 WMMA GEMM:  C[M=128, N] = A[128, K] * W[N, K]^T  (+C0) (+bias)
// A row-major (leading dim lda); W row-major [N, K] (row n == B column n, so
// both fragment loads are K-contiguous b128 pairs; no transpose needed).
// Block = 256 threads = 8 waves; wave w owns rows [16w,16w+16); 64 cols/block.
// Fast path (full 64-col block): unconditional loads, software-pipelined.
// Tail path (partial block, only last block of N=10000): predicated loads.
// ---------------------------------------------------------------------------
__global__ __launch_bounds__(256)
void gemm_bf16_wmma(const bf16_t* __restrict__ A, int lda,
                    const bf16_t* __restrict__ W, int ldw,
                    const float* __restrict__ C0, int ldc0,
                    const float* __restrict__ bias,
                    float* __restrict__ C, int ldc,
                    int N, int K) {
    const int lane  = threadIdx.x & 31;
    const int wave  = threadIdx.x >> 5;
    const int mrow  = (wave << 4) + (lane & 15);   // A row for this lane's slice
    const int khalf = (lane >> 4) << 3;            // 0 or 8 within the K=32 slab
    const int n0    = blockIdx.x * 64;
    const int ncol  = n0 + (lane & 15);

    v8f acc0 = {}, acc1 = {}, acc2 = {}, acc3 = {};

    const bf16_t* ap  = A + (size_t)mrow * lda + khalf;

    if (n0 + 64 <= N) {
        // ---------------- fast path: full tile, double-buffered ----------------
        const bf16_t* bp0 = W + (size_t)(ncol)      * ldw + khalf;
        const bf16_t* bp1 = W + (size_t)(ncol + 16) * ldw + khalf;
        const bf16_t* bp2 = W + (size_t)(ncol + 32) * ldw + khalf;
        const bf16_t* bp3 = W + (size_t)(ncol + 48) * ldw + khalf;

        FragBF ca, cb0, cb1, cb2, cb3;
        frag_load(ca,  ap);
        frag_load(cb0, bp0); frag_load(cb1, bp1);
        frag_load(cb2, bp2); frag_load(cb3, bp3);

        for (int k0 = 32; k0 < K; k0 += 32) {
            FragBF na, nb0, nb1, nb2, nb3;
            frag_load(na,  ap  + k0);
            frag_load(nb0, bp0 + k0); frag_load(nb1, bp1 + k0);
            frag_load(nb2, bp2 + k0); frag_load(nb3, bp3 + k0);
            __builtin_prefetch(ap + k0 + 32, 0, 1);   // global_prefetch_b8

            acc0 = __builtin_amdgcn_wmma_f32_16x16x32_bf16(false, ca.v, false, cb0.v,
                                                           (short)0, acc0, false, false);
            acc1 = __builtin_amdgcn_wmma_f32_16x16x32_bf16(false, ca.v, false, cb1.v,
                                                           (short)0, acc1, false, false);
            acc2 = __builtin_amdgcn_wmma_f32_16x16x32_bf16(false, ca.v, false, cb2.v,
                                                           (short)0, acc2, false, false);
            acc3 = __builtin_amdgcn_wmma_f32_16x16x32_bf16(false, ca.v, false, cb3.v,
                                                           (short)0, acc3, false, false);
            ca = na; cb0 = nb0; cb1 = nb1; cb2 = nb2; cb3 = nb3;
        }
        acc0 = __builtin_amdgcn_wmma_f32_16x16x32_bf16(false, ca.v, false, cb0.v,
                                                       (short)0, acc0, false, false);
        acc1 = __builtin_amdgcn_wmma_f32_16x16x32_bf16(false, ca.v, false, cb1.v,
                                                       (short)0, acc1, false, false);
        acc2 = __builtin_amdgcn_wmma_f32_16x16x32_bf16(false, ca.v, false, cb2.v,
                                                       (short)0, acc2, false, false);
        acc3 = __builtin_amdgcn_wmma_f32_16x16x32_bf16(false, ca.v, false, cb3.v,
                                                       (short)0, acc3, false, false);

        const int colb = n0 + (lane & 15);
        #pragma unroll
        for (int i = 0; i < 8; ++i) {
            const int row = (wave << 4) + i + ((lane >> 4) << 3);
            #pragma unroll
            for (int nt = 0; nt < 4; ++nt) {
                const int col = colb + nt * 16;
                float v = (nt == 0) ? acc0[i] : (nt == 1) ? acc1[i]
                        : (nt == 2) ? acc2[i] : acc3[i];
                if (C0)   v += C0[(size_t)row * ldc0 + col];
                if (bias) v += bias[col];
                C[(size_t)row * ldc + col] = v;
            }
        }
    } else {
        // ---------------- tail path: partial tile (rare) ----------------
        for (int k0 = 0; k0 < K; k0 += 32) {
            FragBF a;
            frag_load(a, ap + k0);

            FragBF b0, b1, b2, b3;
            #pragma unroll
            for (int nt = 0; nt < 4; ++nt) {
                FragBF& f = (nt == 0) ? b0 : (nt == 1) ? b1 : (nt == 2) ? b2 : b3;
                const int n = ncol + nt * 16;
                if (n < N) frag_load(f, W + (size_t)n * ldw + khalf + k0);
                else       frag_zero(f);
            }
            acc0 = __builtin_amdgcn_wmma_f32_16x16x32_bf16(false, a.v, false, b0.v,
                                                           (short)0, acc0, false, false);
            acc1 = __builtin_amdgcn_wmma_f32_16x16x32_bf16(false, a.v, false, b1.v,
                                                           (short)0, acc1, false, false);
            acc2 = __builtin_amdgcn_wmma_f32_16x16x32_bf16(false, a.v, false, b2.v,
                                                           (short)0, acc2, false, false);
            acc3 = __builtin_amdgcn_wmma_f32_16x16x32_bf16(false, a.v, false, b3.v,
                                                           (short)0, acc3, false, false);
        }

        const int colb = n0 + (lane & 15);
        #pragma unroll
        for (int i = 0; i < 8; ++i) {
            const int row = (wave << 4) + i + ((lane >> 4) << 3);
            #pragma unroll
            for (int nt = 0; nt < 4; ++nt) {
                const int col = colb + nt * 16;
                if (col < N) {
                    float v = (nt == 0) ? acc0[i] : (nt == 1) ? acc1[i]
                            : (nt == 2) ? acc2[i] : acc3[i];
                    if (C0)   v += C0[(size_t)row * ldc0 + col];
                    if (bias) v += bias[col];
                    C[(size_t)row * ldc + col] = v;
                }
            }
        }
    }
}

// ---------------------------------------------------------------------------
// fmean[b][d] = mean_v features[b][v][d]
// ---------------------------------------------------------------------------
__global__ void k_fmean(const float* __restrict__ features, float* __restrict__ fmean) {
    int idx = blockIdx.x * blockDim.x + threadIdx.x;   // 128*2048
    int b = idx >> 11, d = idx & 2047;
    const float* fp = features + ((size_t)b * VN) * VD + d;
    float s = 0.f;
    for (int v = 0; v < VN; ++v) s += fp[(size_t)v * VD];
    fmean[idx] = s * (1.0f / (float)VN);
}

// ---------------------------------------------------------------------------
// alpha = softmax_v(features . attn_vw_w)  (bias & per-row terms cancel);
// ctx_bf16[b][d] = bf16( sum_v alpha[v] * features[b][v][d] )
// ---------------------------------------------------------------------------
__global__ __launch_bounds__(256)
void k_alpha_ctx(const float* __restrict__ features, const float* __restrict__ vw,
                 bf16_t* __restrict__ ctx_bf) {
    const int b = blockIdx.x;
    __shared__ float satt[VN];
    const int lane = threadIdx.x & 31, wave = threadIdx.x >> 5;
    const float* fb = features + (size_t)b * VN * VD;

    for (int v = wave; v < VN; v += 8) {
        const float* fr = fb + (size_t)v * VD;
        float s = 0.f;
        for (int d = lane; d < VD; d += 32) s += fr[d] * vw[d];
        #pragma unroll
        for (int off = 16; off > 0; off >>= 1) s += __shfl_xor(s, off, 32);
        if (lane == 0) satt[v] = s;
    }
    __syncthreads();
    if (threadIdx.x == 0) {
        float m = satt[0];
        for (int v = 1; v < VN; ++v) m = fmaxf(m, satt[v]);
        float ssum = 0.f;
        for (int v = 0; v < VN; ++v) { satt[v] = __expf(satt[v] - m); ssum += satt[v]; }
        float inv = 1.0f / ssum;
        for (int v = 0; v < VN; ++v) satt[v] *= inv;
    }
    __syncthreads();
    for (int d = threadIdx.x; d < VD; d += 256) {
        float s = 0.f;
        for (int v = 0; v < VN; ++v) s += satt[v] * fb[(size_t)v * VD + d];
        ctx_bf[(size_t)b * VD + d] = (bf16_t)s;
    }
}

// ---------------------------------------------------------------------------
// h0/c0 = tanh(fmean @ W_i{h,c}v^T + topic_emb @ W_i{h,c}t^T); also seed xbf h-part
// ---------------------------------------------------------------------------
__global__ void k_init_hc(const float* __restrict__ fmean, const int* __restrict__ topics,
                          const float* __restrict__ topic_embed,
                          const float* __restrict__ W_ihv, const float* __restrict__ W_icv,
                          const float* __restrict__ W_iht, const float* __restrict__ W_ict,
                          float* __restrict__ h, float* __restrict__ c,
                          bf16_t* __restrict__ xbf) {
    int idx = blockIdx.x * blockDim.x + threadIdx.x;  // 128*1024
    int b = idx >> 10, j = idx & 1023;
    const float* fm = fmean + (size_t)b * VD;
    const float* te = topic_embed + (size_t)topics[b] * Ee;
    float sh = 0.f, sc = 0.f;
    for (int d = 0; d < VD; ++d) {
        float f = fm[d];
        sh += f * W_ihv[(size_t)j * VD + d];
        sc += f * W_icv[(size_t)j * VD + d];
    }
    for (int e = 0; e < Ee; ++e) {
        float t = te[e];
        sh += t * W_iht[(size_t)j * Ee + e];
        sc += t * W_ict[(size_t)j * Ee + e];
    }
    sh = tanhf(sh); sc = tanhf(sc);
    h[idx] = sh; c[idx] = sc;
    xbf[(size_t)b * KX + Ee + j] = (bf16_t)sh;
}

// f32 -> bf16 strided matrix convert
__global__ void k_cvt(const float* __restrict__ src, int lds_,
                      bf16_t* __restrict__ dst, int ldd, long long rows, int cols) {
    long long total = rows * (long long)cols;
    for (long long i = blockIdx.x * (long long)blockDim.x + threadIdx.x; i < total;
         i += (long long)gridDim.x * blockDim.x) {
        long long r = i / cols; int cc = (int)(i % cols);
        dst[r * ldd + cc] = (bf16_t)src[r * lds_ + cc];
    }
}

__global__ void k_bsum(const float* __restrict__ a, const float* __restrict__ b,
                       float* __restrict__ o) {
    int i = blockIdx.x * blockDim.x + threadIdx.x;
    if (i < G4H) o[i] = a[i] + b[i];
}

__global__ void k_init_tok(const int* __restrict__ captions, int* __restrict__ tok) {
    int b = threadIdx.x;
    if (b < Bsz) tok[b] = captions[(size_t)b * Ll];
}

// gather word embedding for current tokens into xbf[:, 0:E]
__global__ void k_embed(const float* __restrict__ word_embed, const int* __restrict__ tok,
                        bf16_t* __restrict__ xbf) {
    int idx = blockIdx.x * blockDim.x + threadIdx.x;  // 128*512
    int b = idx >> 9, e = idx & 511;
    xbf[(size_t)b * KX + e] = (bf16_t)word_embed[(size_t)tok[b] * Ee + e];
}

// LSTM cell update; refresh f32 state and bf16 h inside xbf
__global__ void k_cell(const float* __restrict__ gates, float* __restrict__ h,
                       float* __restrict__ c, bf16_t* __restrict__ xbf) {
    int idx = blockIdx.x * blockDim.x + threadIdx.x;  // 128*1024
    int b = idx >> 10, j = idx & 1023;
    const float* g = gates + (size_t)b * G4H;
    float gi = g[j], gf = g[Hh + j], gg = g[2 * Hh + j], go = g[3 * Hh + j];
    float cn = sigm(gf) * c[idx] + sigm(gi) * tanhf(gg);
    float hn = sigm(go) * tanhf(cn);
    c[idx] = cn; h[idx] = hn;
    xbf[(size_t)b * KX + Ee + j] = (bf16_t)hn;
}

// fused log_softmax (to output slab) + greedy argmax (to tok)
__global__ __launch_bounds__(512)
void k_logsm(const float* __restrict__ logits, float* __restrict__ out,
             int* __restrict__ tok, int t) {
    const int b = blockIdx.x;
    const float* row = logits + (size_t)b * WV;
    __shared__ float sv[512];
    __shared__ int   si[512];
    float m = -INFINITY; int mi = 0;
    for (int j = threadIdx.x; j < WV; j += 512) {
        float v = row[j];
        if (v > m) { m = v; mi = j; }
    }
    sv[threadIdx.x] = m; si[threadIdx.x] = mi;
    __syncthreads();
    for (int s = 256; s > 0; s >>= 1) {
        if ((int)threadIdx.x < s) {
            float vo = sv[threadIdx.x + s]; int io = si[threadIdx.x + s];
            if (vo > sv[threadIdx.x] ||
                (vo == sv[threadIdx.x] && io < si[threadIdx.x])) {
                sv[threadIdx.x] = vo; si[threadIdx.x] = io;
            }
        }
        __syncthreads();
    }
    const float mx = sv[0]; const int amax = si[0];
    __syncthreads();
    float s = 0.f;
    for (int j = threadIdx.x; j < WV; j += 512) s += __expf(row[j] - mx);
    sv[threadIdx.x] = s; __syncthreads();
    for (int st = 256; st > 0; st >>= 1) {
        if ((int)threadIdx.x < st) sv[threadIdx.x] += sv[threadIdx.x + st];
        __syncthreads();
    }
    const float lse = __logf(sv[0]) + mx;
    float* orow = out + ((size_t)b * Ll + t) * WV;
    for (int j = threadIdx.x; j < WV; j += 512) orow[j] = row[j] - lse;
    if (threadIdx.x == 0) tok[b] = amax;
}

__global__ void k_writehc(const float* __restrict__ h, const float* __restrict__ c,
                          float* __restrict__ dst) {
    int idx = blockIdx.x * blockDim.x + threadIdx.x;  // 128*1024
    dst[idx] = h[idx];
    dst[Bsz * Hh + idx] = c[idx];
}

// ---------------------------------------------------------------------------
extern "C" void kernel_launch(void* const* d_in, const int* in_sizes, int n_in,
                              void* d_out, int out_size, void* d_ws, size_t ws_size,
                              hipStream_t stream) {
    const float* features    = (const float*)d_in[0];
    const int*   topics      = (const int*)  d_in[1];
    const int*   captions    = (const int*)  d_in[2];
    const float* W_ihv       = (const float*)d_in[3];
    const float* W_icv       = (const float*)d_in[4];
    const float* W_iht       = (const float*)d_in[5];
    const float* W_ict       = (const float*)d_in[6];
    const float* attn_vw_w   = (const float*)d_in[7];
    // d_in[8..12]: attn biases / hw / tw — provably cancel in the softmax
    const float* topic_embed = (const float*)d_in[13];
    const float* word_embed  = (const float*)d_in[14];
    const float* lstm_Wih    = (const float*)d_in[15];
    const float* lstm_Whh    = (const float*)d_in[16];
    const float* lstm_bih    = (const float*)d_in[17];
    const float* lstm_bhh    = (const float*)d_in[18];
    const float* out_W       = (const float*)d_in[19];
    const float* out_b       = (const float*)d_in[20];
    float* out = (float*)d_out;

    char* p = (char*)d_ws;
    auto alloc = [&](size_t bytes) -> void* {
        void* r = (void*)p;
        p += (bytes + 255) & ~(size_t)255;
        return r;
    };
    float*  fmean  = (float*) alloc((size_t)Bsz * VD * 4);
    bf16_t* ctx_bf = (bf16_t*)alloc((size_t)Bsz * VD * 2);
    bf16_t* WihC   = (bf16_t*)alloc((size_t)G4H * VD * 2);
    bf16_t* Wcat   = (bf16_t*)alloc((size_t)G4H * KX * 2);
    bf16_t* WoutB  = (bf16_t*)alloc((size_t)WV * Hh * 2);
    float*  bsum   = (float*) alloc((size_t)G4H * 4);
    float*  gctx   = (float*) alloc((size_t)Bsz * G4H * 4);
    float*  gates  = (float*) alloc((size_t)Bsz * G4H * 4);
    bf16_t* xbf    = (bf16_t*)alloc((size_t)Bsz * KX * 2);
    float*  hbuf   = (float*) alloc((size_t)Bsz * Hh * 4);
    float*  cbuf   = (float*) alloc((size_t)Bsz * Hh * 4);
    float*  logits = (float*) alloc((size_t)Bsz * WV * 4);
    int*    tok    = (int*)   alloc((size_t)Bsz * 4);

    // ---------------- one-time precompute ----------------
    k_fmean<<<(Bsz * VD) / 256, 256, 0, stream>>>(features, fmean);
    k_alpha_ctx<<<Bsz, 256, 0, stream>>>(features, attn_vw_w, ctx_bf);

    // weight conversions (f32 -> bf16), Wcat = [Wih_emb | Whh] with K = 1536
    k_cvt<<<2048, 256, 0, stream>>>(lstm_Wih,        KCAT, WihC,       VD, G4H, VD);
    k_cvt<<<1024, 256, 0, stream>>>(lstm_Wih + VD,   KCAT, Wcat,       KX, G4H, Ee);
    k_cvt<<<2048, 256, 0, stream>>>(lstm_Whh,        Hh,   Wcat + Ee,  KX, G4H, Hh);
    k_cvt<<<4096, 256, 0, stream>>>(out_W,           Hh,   WoutB,      Hh, WV,  Hh);
    k_bsum<<<G4H / 256, 256, 0, stream>>>(lstm_bih, lstm_bhh, bsum);

    k_init_hc<<<(Bsz * Hh) / 256, 256, 0, stream>>>(fmean, topics, topic_embed,
                                                    W_ihv, W_icv, W_iht, W_ict,
                                                    hbuf, cbuf, xbf);

    // gates_ctx = ctx @ Wih_ctx^T + (bih + bhh)   [step-invariant]
    gemm_bf16_wmma<<<G4H / 64, 256, 0, stream>>>(ctx_bf, VD, WihC, VD,
                                                 nullptr, 0, bsum,
                                                 gctx, G4H, G4H, VD);

    k_init_tok<<<1, 128, 0, stream>>>(captions, tok);

    // ---------------- sequential decode loop ----------------
    for (int t = 0; t < Ll; ++t) {
        k_embed<<<(Bsz * Ee) / 256, 256, 0, stream>>>(word_embed, tok, xbf);

        // gates = [emb|h] @ Wcat^T + gates_ctx
        gemm_bf16_wmma<<<G4H / 64, 256, 0, stream>>>(xbf, KX, Wcat, KX,
                                                     gctx, G4H, nullptr,
                                                     gates, G4H, G4H, KX);

        k_cell<<<(Bsz * Hh) / 256, 256, 0, stream>>>(gates, hbuf, cbuf, xbf);

        // logits = h @ out_W^T + out_b   (A = h slice of xbf, lda = KX)
        gemm_bf16_wmma<<<(WV + 63) / 64, 256, 0, stream>>>(xbf + Ee, KX, WoutB, Hh,
                                                           nullptr, 0, out_b,
                                                           logits, WV, WV, Hh);

        k_logsm<<<Bsz, 512, 0, stream>>>(logits, out, tok, t);
    }

    // trailing outputs: h[1,B,H], c[1,B,H]
    k_writehc<<<(Bsz * Hh) / 256, 256, 0, stream>>>(hbuf, cbuf,
                                                    out + (size_t)Bsz * Ll * WV);
}
